// MaskNet_15298673508509
// MI455X (gfx1250) — compile-verified, matrix operand order checked by reading, not run on hardware
//
#include <hip/hip_runtime.h>
#include <hip/hip_bf16.h>
#include <math.h>

typedef __attribute__((ext_vector_type(16))) _Float16 v16h;
typedef __attribute__((ext_vector_type(8)))  _Float16 v8h;
typedef __attribute__((ext_vector_type(8)))  float    v8f;

#define RES 7
#define NPOS 64
#define RROI 512
#define CF  256
#define HF  36
#define WF  36
#define NTOT (RROI * NPOS)   // 32768 padded positions

// ---------------------------------------------------------------------------
// Fragment helper: build a 16-half WMMA operand from two 16B chunks.
// 16-bit A (16x32) layout: lane L<16 -> row M=L, halfs K in {g..g+7, g+16..g+23}
// with g = (lane<16 ? 0 : 8).  B (32x16) uses the mirrored layout per column.
// ---------------------------------------------------------------------------
__device__ __forceinline__ v16h frag16(const _Float16* p0, const _Float16* p1) {
    v8h a = *(const v8h*)p0;
    v8h b = *(const v8h*)p1;
    v16h r;
#pragma unroll
    for (int i = 0; i < 8; ++i) { r[i] = a[i]; r[i + 8] = b[i]; }
    return r;
}

__device__ __forceinline__ v8f wmma32(v16h a, v16h b, v8f c) {
    return __builtin_amdgcn_wmma_f32_16x16x32_f16(false, a, false, b,
                                                  (short)0, c, false, false);
}

// ---------------------------------------------------------------------------
// CDNA5 async global->LDS copy (ASYNCcnt-tracked DMA, no VGPR round trip).
// LDS address = low 32 bits of the shared-aperture flat address (ISA 10.2).
// 16 bytes per active lane.
// ---------------------------------------------------------------------------
__device__ __forceinline__ void async_copy16(void* lds_ptr, const void* gptr) {
    unsigned lds_addr = (unsigned)(size_t)lds_ptr;
    asm volatile("global_load_async_to_lds_b128 %0, %1, off"
                 :: "v"(lds_addr), "v"(gptr)
                 : "memory");
}
__device__ __forceinline__ void async_wait0() {
    asm volatile("s_wait_asynccnt 0" ::: "memory");
}

// ---------------------------------------------------------------------------
// Weight prep: fp32 -> f16, GEMM-friendly layouts
// ---------------------------------------------------------------------------
__global__ void cast_f16_kernel(const float* __restrict__ src,
                                _Float16* __restrict__ dst, int n) {
    int i = blockIdx.x * blockDim.x + threadIdx.x;
    if (i < n) dst[i] = (_Float16)src[i];
}

// w2: [3][co=128][ci=128][3][3] -> W2h [3][tap=9][co][ci]
__global__ void prep_w2_kernel(const float* __restrict__ src,
                               _Float16* __restrict__ dst) {
    int i = blockIdx.x * blockDim.x + threadIdx.x;
    const int n = 3 * 9 * 128 * 128;
    if (i >= n) return;
    int ci  = i & 127;
    int co  = (i >> 7) & 127;
    int tap = (i >> 14) % 9;
    int blk = i / (9 * 128 * 128);
    dst[i] = (_Float16)src[(((blk * 128 + co) * 128 + ci) * 9) + tap];
}

// wt: [ci=256][co=256][2][2] -> WTh [d=4][co][ci]
__global__ void prep_wt_kernel(const float* __restrict__ src,
                               _Float16* __restrict__ dst) {
    int i = blockIdx.x * blockDim.x + threadIdx.x;
    const int n = 4 * 256 * 256;
    if (i >= n) return;
    int ci = i & 255;
    int co = (i >> 8) & 255;
    int d  = i >> 16;
    dst[i] = (_Float16)src[(ci * 256 + co) * 4 + d];
}

// ---------------------------------------------------------------------------
// PrRoI pooling: bin weights (exact bilinear-kernel integral), then pooling
// ---------------------------------------------------------------------------
__device__ __forceinline__ float Fk(float t) {
    t = fminf(fmaxf(t, -1.0f), 1.0f);
    return (t < 0.0f) ? 0.5f * (t + 1.0f) * (t + 1.0f)
                      : 1.0f - 0.5f * (1.0f - t) * (1.0f - t);
}

__global__ void binweights_kernel(const float* __restrict__ bbox,
                                  float* __restrict__ Wx,
                                  float* __restrict__ Wy,
                                  float* __restrict__ inv_area) {
    const float S = 1.0f / 16.0f;
    int r = blockIdx.x;
    float x1 = bbox[r * 4 + 0] * S;
    float y1 = bbox[r * 4 + 1] * S;
    float bwx = (bbox[r * 4 + 2] * S) / (float)RES;
    float bwy = (bbox[r * 4 + 3] * S) / (float)RES;
    for (int e = threadIdx.x; e < 2 * RES * WF; e += blockDim.x) {
        int isY = e >= RES * WF;
        int ee  = isY ? e - RES * WF : e;
        int p   = ee / WF;
        int pix = ee % WF;
        float lo = isY ? y1 : x1;
        float bw = isY ? bwy : bwx;
        float s = lo + bw * (float)p;
        float v = Fk(s + bw - (float)pix) - Fk(s - (float)pix);
        (isY ? Wy : Wx)[r * RES * WF + ee] = v;
    }
    if (threadIdx.x == 0) {
        float area = bwx * bwy;
        inv_area[r] = (area > 0.0f) ? 1.0f / fmaxf(area, 1e-12f) : 0.0f;
    }
}

// One block per ROI, one thread per channel. Output f16 act0[pos][256],
// positions padded 49->64 with zeros.
__global__ void __launch_bounds__(256)
prroi_kernel(const float* __restrict__ feat,
             const float* __restrict__ Wx,
             const float* __restrict__ Wy,
             const float* __restrict__ inv_area,
             _Float16* __restrict__ act) {
    __shared__ float sWx[RES * WF];
    __shared__ float sWy[RES * HF];
    int r = blockIdx.x;
    int c = threadIdx.x;
    for (int e = c; e < RES * WF; e += blockDim.x) {
        sWx[e] = Wx[r * RES * WF + e];
        sWy[e] = Wy[r * RES * HF + e];
    }
    __syncthreads();

    int b = r >> 5;  // 32 boxes per frame
    const float* F = feat + ((size_t)(b * CF + c)) * (HF * WF);

    float acc[49];
#pragma unroll
    for (int i = 0; i < 49; ++i) acc[i] = 0.0f;

    for (int h = 0; h < HF; ++h) {
        float rowv[WF];
#pragma unroll
        for (int w = 0; w < WF; ++w) rowv[w] = F[h * WF + w];
        float sx[RES];
#pragma unroll
        for (int q = 0; q < RES; ++q) {
            float s = 0.0f;
#pragma unroll
            for (int w = 0; w < WF; ++w) s += sWx[q * WF + w] * rowv[w];
            sx[q] = s;
        }
#pragma unroll
        for (int p = 0; p < RES; ++p) {
            float wy = sWy[p * HF + h];
#pragma unroll
            for (int q = 0; q < RES; ++q) acc[p * RES + q] += wy * sx[q];
        }
    }
    float ia = inv_area[r];
    size_t base = (size_t)r * NPOS;
    for (int idx = 0; idx < 49; ++idx)
        act[(base + idx) * CF + c] = (_Float16)(acc[idx] * ia);
    for (int idx = 49; idx < NPOS; ++idx)
        act[(base + idx) * CF + c] = (_Float16)0.0f;
}

// ---------------------------------------------------------------------------
// 1x1 conv + frozen-BN (+ optional residual) + ReLU as per-ROI WMMA GEMM.
// act: [NTOT][Cin] f16, W: [Cout][Cin] f16, out: [NTOT][Cout] f16.
// Block = 1 ROI, 4 waves (one N-tile of 16 positions each).
// ---------------------------------------------------------------------------
__global__ void __launch_bounds__(128)
gemm1x1_kernel(const _Float16* __restrict__ act,
               const _Float16* __restrict__ W,
               const float* __restrict__ scale,
               const float* __restrict__ bias,
               const _Float16* __restrict__ resid,  // [NTOT][Cout] or null
               _Float16* __restrict__ out,
               int Cin, int Cout) {
    __shared__ _Float16 xs[NPOS * 256];
    int r = blockIdx.x;
    int tid = threadIdx.x;
    size_t base = (size_t)r * NPOS;

    // async-DMA the ROI's input tile into LDS: xs[pos][ci]
    int cg8 = Cin >> 3;
    for (int e = tid; e < NPOS * cg8; e += blockDim.x) {
        int pos = e / cg8, cg = e - pos * cg8;
        async_copy16(&xs[pos * Cin + cg * 8],
                     &act[(base + pos) * Cin + cg * 8]);
    }
    async_wait0();
    __syncthreads();

    int wave = tid >> 5, lane = tid & 31;
    int n = lane & 15, hi = lane >> 4, g = hi * 8;
    int col = wave * 16 + n;
    const _Float16* xrow = &xs[col * Cin];

    for (int mt = 0; mt < (Cout >> 4); ++mt) {
        v8f acc = {};
        const _Float16* arow = &W[(size_t)(mt * 16 + n) * Cin];
        if (mt + 1 < (Cout >> 4))
            __builtin_prefetch(&W[(size_t)((mt + 1) * 16 + n) * Cin], 0, 0);
        for (int kc = 0; kc < Cin; kc += 32) {
            v16h a = frag16(arow + kc + g, arow + kc + 16 + g);
            v16h b = frag16(xrow + kc + g, xrow + kc + 16 + g);
            acc = wmma32(a, b, acc);
        }
        // C layout: lane<16 -> (M=j, N=lane); lane>=16 -> (M=j+8, N=lane-16)
        size_t pos = base + col;
        int co0 = mt * 16 + hi * 8;
        v8h o;
#pragma unroll
        for (int j = 0; j < 8; ++j) {
            int co = co0 + j;
            float v = acc[j] * scale[co] + bias[co];
            if (resid) v += (float)resid[pos * Cout + co];
            o[j] = (_Float16)fmaxf(v, 0.0f);
        }
        *(v8h*)&out[pos * Cout + co0] = o;
    }
}

// ---------------------------------------------------------------------------
// 3x3 conv (pad 1) + BN + ReLU, implicit im2col over 9 taps. Cin=Cout=128.
// W: [9][128][128]. Block = 1 ROI, 8 waves: wave -> (ntile, m-half).
// ---------------------------------------------------------------------------
__global__ void __launch_bounds__(256)
gemm3x3_kernel(const _Float16* __restrict__ act,   // [NTOT][128]
               const _Float16* __restrict__ W,     // [9][128][128]
               const float* __restrict__ scale,
               const float* __restrict__ bias,
               _Float16* __restrict__ out) {       // [NTOT][128]
    const int C = 128;
    __shared__ _Float16 xs[NPOS * 128];
    int r = blockIdx.x;
    int tid = threadIdx.x;
    size_t base = (size_t)r * NPOS;

    for (int e = tid; e < NPOS * (C >> 3); e += blockDim.x) {
        int pos = e >> 4, cg = e & 15;
        async_copy16(&xs[pos * C + cg * 8],
                     &act[(base + pos) * C + cg * 8]);
    }
    async_wait0();
    __syncthreads();

    int wave = tid >> 5, lane = tid & 31;
    int n = lane & 15, hi = lane >> 4, g = hi * 8;
    int ntile = wave & 3;
    int mbase = (wave >> 2) * 4;          // 8 m-tiles split across 2 groups
    int idx = ntile * 16 + n;             // output position within ROI
    int p = idx / RES, q = idx - p * RES;
    bool ovalid = idx < 49;

    v8f acc[4];
#pragma unroll
    for (int i = 0; i < 4; ++i) acc[i] = (v8f){};

    for (int t = 0; t < 9; ++t) {
        int dy = t / 3 - 1, dx = t % 3 - 1;
        int sp = p + dy, sq = q + dx;
        bool ok = ovalid && (unsigned)sp < RES && (unsigned)sq < RES;
        int sidx = sp * RES + sq;
        const _Float16* Wt = W + (size_t)t * C * C;
        for (int kc = 0; kc < C; kc += 32) {
            v16h b = {};
            if (ok) b = frag16(&xs[sidx * C + kc + g],
                               &xs[sidx * C + kc + 16 + g]);
#pragma unroll
            for (int mi = 0; mi < 4; ++mi) {
                const _Float16* arow = &Wt[(size_t)((mbase + mi) * 16 + n) * C];
                v16h a = frag16(arow + kc + g, arow + kc + 16 + g);
                acc[mi] = wmma32(a, b, acc[mi]);
            }
        }
    }
    size_t pos = base + idx;
#pragma unroll
    for (int mi = 0; mi < 4; ++mi) {
        int co0 = (mbase + mi) * 16 + hi * 8;
        v8h o;
#pragma unroll
        for (int j = 0; j < 8; ++j) {
            int co = co0 + j;
            o[j] = (_Float16)fmaxf(acc[mi][j] * scale[co] + bias[co], 0.0f);
        }
        *(v8h*)&out[pos * C + co0] = o;
    }
}

// ---------------------------------------------------------------------------
// Fused 2x2-stride-2 deconv (+bt, ReLU) + 1x1 mask conv + sigmoid.
// 4 sub-pixel GEMMs [256,256]x[256,64]; the channel dot with wm is reduced
// in-wave (mtile loop + shfl_xor across the two half-lane groups), so the
// 14x14 deconv activation is never materialized.
// ---------------------------------------------------------------------------
__global__ void __launch_bounds__(128)
deconv_mask_kernel(const _Float16* __restrict__ act,  // [NTOT][256]
                   const _Float16* __restrict__ WT,   // [4][256][256]
                   const float* __restrict__ bt,      // [256]
                   const float* __restrict__ wm,      // [256]
                   const float* __restrict__ bm,      // [1]
                   float* __restrict__ out) {         // [512][196]
    const int C = 256;
    __shared__ _Float16 xs[NPOS * 256];
    int r = blockIdx.x;
    int tid = threadIdx.x;
    size_t base = (size_t)r * NPOS;

    for (int e = tid; e < NPOS * (C >> 3); e += blockDim.x) {
        int pos = e >> 5, cg = e & 31;
        async_copy16(&xs[pos * C + cg * 8],
                     &act[(base + pos) * C + cg * 8]);
    }
    async_wait0();
    __syncthreads();

    int wave = tid >> 5, lane = tid & 31;
    int n = lane & 15, hi = lane >> 4, g = hi * 8;
    float bmv = bm[0];

    for (int combo = wave; combo < 16; combo += 4) {
        int d = combo >> 2;        // sub-pixel (dy,dx)
        int ntile = combo & 3;
        int idx = ntile * 16 + n;  // input 7x7 position
        const _Float16* xrow = &xs[idx * C];
        const _Float16* Wd = WT + (size_t)d * C * C;

        float lanesum = 0.0f;
        for (int mt = 0; mt < 16; ++mt) {
            v8f acc = {};
            const _Float16* arow = &Wd[(size_t)(mt * 16 + n) * C];
            if (mt + 1 < 16)
                __builtin_prefetch(&Wd[(size_t)((mt + 1) * 16 + n) * C], 0, 0);
            for (int kc = 0; kc < C; kc += 32) {
                v16h a = frag16(arow + kc + g, arow + kc + 16 + g);
                v16h b = frag16(xrow + kc + g, xrow + kc + 16 + g);
                acc = wmma32(a, b, acc);
            }
            int co0 = mt * 16 + hi * 8;
#pragma unroll
            for (int j = 0; j < 8; ++j) {
                int co = co0 + j;
                float v = fmaxf(acc[j] + bt[co], 0.0f);
                lanesum += v * wm[co];
            }
        }
        // columns n and n+16 hold co-halves of the same position: combine
        float total = lanesum + __shfl_xor(lanesum, 16, 32);
        if (hi == 0 && idx < 49) {
            int p = idx / RES, q = idx - p * RES;
            int dy = d >> 1, dx = d & 1;
            int opix = (2 * p + dy) * 14 + (2 * q + dx);
            float z = total + bmv;
            out[r * 196 + opix] = 1.0f / (1.0f + __expf(-z));
        }
    }
}

// ---------------------------------------------------------------------------
// Host-side orchestration
// ---------------------------------------------------------------------------
extern "C" void kernel_launch(void* const* d_in, const int* in_sizes, int n_in,
                              void* d_out, int out_size, void* d_ws, size_t ws_size,
                              hipStream_t stream) {
    (void)in_sizes; (void)n_in; (void)out_size; (void)ws_size;
    const float* feat = (const float*)d_in[0];
    const float* bbox = (const float*)d_in[1];
    const float* w1   = (const float*)d_in[2];
    const float* s1   = (const float*)d_in[3];
    const float* b1   = (const float*)d_in[4];
    const float* w2   = (const float*)d_in[5];
    const float* s2   = (const float*)d_in[6];
    const float* b2   = (const float*)d_in[7];
    const float* w3   = (const float*)d_in[8];
    const float* s3   = (const float*)d_in[9];
    const float* b3   = (const float*)d_in[10];
    const float* wt   = (const float*)d_in[11];
    const float* bt   = (const float*)d_in[12];
    const float* wm   = (const float*)d_in[13];
    const float* bm   = (const float*)d_in[14];
    float* outp = (float*)d_out;

    char* ws = (char*)d_ws;
    size_t off = 0;
    auto take = [&](size_t bytes) { char* p = ws + off; off += (bytes + 255) & ~(size_t)255; return p; };

    float*     Wx   = (float*)take((size_t)RROI * RES * WF * 4);
    float*     Wy   = (float*)take((size_t)RROI * RES * HF * 4);
    float*     inv  = (float*)take((size_t)RROI * 4);
    _Float16*  W1h  = (_Float16*)take((size_t)3 * 128 * 256 * 2);
    _Float16*  W2h  = (_Float16*)take((size_t)3 * 9 * 128 * 128 * 2);
    _Float16*  W3h  = (_Float16*)take((size_t)3 * 256 * 128 * 2);
    _Float16*  WTh  = (_Float16*)take((size_t)4 * 256 * 256 * 2);
    _Float16*  bufA = (_Float16*)take((size_t)NTOT * 256 * 2);
    _Float16*  bufD = (_Float16*)take((size_t)NTOT * 256 * 2);
    _Float16*  bufB = (_Float16*)take((size_t)NTOT * 128 * 2);
    _Float16*  bufC = (_Float16*)take((size_t)NTOT * 128 * 2);

    // weight prep
    {
        int n1 = 3 * 128 * 256;
        cast_f16_kernel<<<(n1 + 255) / 256, 256, 0, stream>>>(w1, W1h, n1);
        int n3 = 3 * 256 * 128;
        cast_f16_kernel<<<(n3 + 255) / 256, 256, 0, stream>>>(w3, W3h, n3);
        int n2 = 3 * 9 * 128 * 128;
        prep_w2_kernel<<<(n2 + 255) / 256, 256, 0, stream>>>(w2, W2h);
        int nt = 4 * 256 * 256;
        prep_wt_kernel<<<(nt + 255) / 256, 256, 0, stream>>>(wt, WTh);
    }

    // PrRoI pooling -> bufA [NTOT][256] f16 (padded 49->64)
    binweights_kernel<<<RROI, 128, 0, stream>>>(bbox, Wx, Wy, inv);
    prroi_kernel<<<RROI, 256, 0, stream>>>(feat, Wx, Wy, inv, bufA);

    // 3 bottleneck blocks (ping-pong bufA <-> bufD)
    _Float16* xin = bufA;
    _Float16* xout = bufD;
    for (int i = 0; i < 3; ++i) {
        gemm1x1_kernel<<<RROI, 128, 0, stream>>>(
            xin, W1h + (size_t)i * 128 * 256, s1 + i * 128, b1 + i * 128,
            nullptr, bufB, 256, 128);
        gemm3x3_kernel<<<RROI, 256, 0, stream>>>(
            bufB, W2h + (size_t)i * 9 * 128 * 128, s2 + i * 128, b2 + i * 128,
            bufC);
        gemm1x1_kernel<<<RROI, 128, 0, stream>>>(
            bufC, W3h + (size_t)i * 256 * 128, s3 + i * 256, b3 + i * 256,
            xin /*residual*/, xout, 128, 256);
        _Float16* t = xin; xin = xout; xout = t;
    }

    // fused deconv + mask conv + sigmoid -> d_out [512*196] fp32
    deconv_mask_kernel<<<RROI, 128, 0, stream>>>(xin, WTh, bt, wm, bm, outp);
}
